// RNNModule_33930241639041
// MI455X (gfx1250) — matrix operand order repaired and checked by reading, not compiled, not run on hardware
//
#include <hip/hip_runtime.h>

#define B_ 128
#define L_ 256
#define D_ 512
#define H_ 1024

typedef __attribute__((ext_vector_type(16))) __bf16 v16bf;
typedef __attribute__((ext_vector_type(8)))  float  v8f;

struct __align__(16) U4 { unsigned int x, y, z, w; };

union FragU {
  U4 q[2];
  unsigned int u[8];
  v16bf bf;
};

__device__ __forceinline__ unsigned short f2bf(float f) {
  unsigned int x = __float_as_uint(f);
  return (unsigned short)((x + 0x7FFFu + ((x >> 16) & 1u)) >> 16);
}
__device__ __forceinline__ unsigned int pack2bf(float lo, float hi) {
  return (unsigned int)f2bf(lo) | ((unsigned int)f2bf(hi) << 16);
}
__device__ __forceinline__ float bf2f_lo(unsigned int u) { return __uint_as_float(u << 16); }
__device__ __forceinline__ float bf2f_hi(unsigned int u) { return __uint_as_float(u & 0xFFFF0000u); }
__device__ __forceinline__ float sigmoidf_(float z) { return 1.0f / (1.0f + __expf(-z)); }

// ---------------------------------------------------------------------------
// Pack Wh (H x H) and Wx (D x H) fp32 -> bf16 WMMA B-fragment layout.
// Fragment (ktile, ntile) = 32 lanes x 8 dwords, stored as 2 chunks of
// 32 x uint4 so the consumer does two coalesced b128 loads per lane.
// B-matrix 32x16 layout: lane<16 -> n = n0+lane, K = k0+0..7, k0+16..23; lane>=16 ->
// n = n0+lane-16, K = k0+8..15, k0+24..31
// ---------------------------------------------------------------------------
__global__ void rnn_prep_frags(const float* __restrict__ Wh,
                               const float* __restrict__ Wx,
                               U4* __restrict__ whb, U4* __restrict__ wxb) {
  const int frag = blockIdx.x;
  const int lane = threadIdx.x;  // 32 threads
  const float* W;
  U4* dst;
  int kt, nt;
  if (frag < 32 * 64) {           // Wh: 32 ktiles x 64 ntiles
    kt = frag >> 6; nt = frag & 63;
    W = Wh; dst = whb + (size_t)frag * 64;
  } else {                        // Wx: 16 ktiles x 64 ntiles
    int f = frag - 2048;
    kt = f >> 6; nt = f & 63;
    W = Wx; dst = wxb + (size_t)f * 64;
  }
  const int n  = nt * 16 + (lane & 15);
  const int kb = kt * 32 + ((lane & 16) ? 8 : 0);
  unsigned int d[8];
#pragma unroll
  for (int v = 0; v < 8; ++v) {
    const int k = kb + ((v < 4) ? 2 * v : 16 + 2 * (v - 4));
    d[v] = pack2bf(W[(size_t)k * H_ + n], W[(size_t)(k + 1) * H_ + n]);
  }
  U4 q0 = {d[0], d[1], d[2], d[3]};
  U4 q1 = {d[4], d[5], d[6], d[7]};
  dst[lane] = q0;
  dst[32 + lane] = q1;
}

// ---------------------------------------------------------------------------
// xp = xT @ Wx + bx   (time-major rows r = t*B + b), bf16 WMMA, f32 accum.
// Output stored in C-fragment bf16 layout: frag (rowtile rt, ntile nt) =
// 32 lanes x 1 uint4 (8 bf16), so the scan kernel init is one b128 load.
// Block: 16 rows, 8 waves x 8 ntiles = all 1024 columns. Grid: 2048 blocks.
// ---------------------------------------------------------------------------
__global__ __launch_bounds__(256, 1) void rnn_xp_gemm(
    const float* __restrict__ x, const U4* __restrict__ wxb,
    const float* __restrict__ bx, U4* __restrict__ xpf) {
  __shared__ __align__(16) unsigned short xl[16 * D_];  // 16 KB bf16 rows

  const int lane = threadIdx.x & 31;
  const int wave = threadIdx.x >> 5;
  const int r0 = blockIdx.x * 16;

  // Stage 16 rows of x (t-major) into LDS as bf16, coalesced along D.
  for (int idx = threadIdx.x; idx < 16 * D_; idx += 256) {
    const int i = idx >> 9;          // row in tile
    const int k = idx & (D_ - 1);
    const int r = r0 + i;
    const int t = r >> 7;            // r / B
    const int b = r & (B_ - 1);      // r % B
    xl[i * D_ + k] = f2bf(x[((size_t)b * L_ + t) * D_ + k]);
  }
  __syncthreads();

  const int ncol = lane & 15;
  const int hi4 = (lane & 16) ? 4 : 0;

  v8f acc[8];
#pragma unroll
  for (int j = 0; j < 8; ++j) {
    const float bv = bx[wave * 128 + j * 16 + ncol];
#pragma unroll
    for (int v = 0; v < 8; ++v) acc[j][v] = bv;
  }

  const unsigned int* arow = (const unsigned int*)xl + (lane & 15) * (D_ / 2);
#pragma unroll
  for (int kt = 0; kt < 16; ++kt) {
    FragU a;
    const int d0 = (kt << 4) + hi4;  // (kt*32)/2 + lane-half select
    a.q[0] = *(const U4*)(arow + d0);
    a.q[1] = *(const U4*)(arow + d0 + 8);
#pragma unroll
    for (int j = 0; j < 8; ++j) {
      FragU bfr;
      const U4* fb = wxb + ((size_t)(kt * 64 + wave * 8 + j)) * 64;
      bfr.q[0] = fb[lane];
      bfr.q[1] = fb[32 + lane];
      acc[j] = __builtin_amdgcn_wmma_f32_16x16x32_bf16(
          false, a.bf, false, bfr.bf, (short)0, acc[j], false, false);
    }
  }

  // Store C-fragments as packed bf16 (one b128 per lane per ntile).
#pragma unroll
  for (int j = 0; j < 8; ++j) {
    U4 o;
    o.x = pack2bf(acc[j][0], acc[j][1]);
    o.y = pack2bf(acc[j][2], acc[j][3]);
    o.z = pack2bf(acc[j][4], acc[j][5]);
    o.w = pack2bf(acc[j][6], acc[j][7]);
    xpf[((size_t)blockIdx.x * 64 + (wave * 8 + j)) * 32 + lane] = o;
  }
}

// ---------------------------------------------------------------------------
// Sequential scan: 8 workgroups, each owns a 16-row batch tile.
// Per step: z = h@Wh + bh + xp[t] (bf16 WMMA from LDS-resident h), then
// elementwise LSTM-like update with c state held in VGPRs, h written back to
// LDS for the next step, and the O=2 output projection fused via ds_add_f32.
// ---------------------------------------------------------------------------
__global__ __launch_bounds__(256, 1) void rnn_scan(
    const U4* __restrict__ whb, const U4* __restrict__ xpf,
    const float* __restrict__ bh, const float* __restrict__ Wo,
    const float* __restrict__ bo, const int* __restrict__ slen,
    float* __restrict__ out) {
  __shared__ __align__(16) unsigned short hl[16 * H_];  // 32 KB bf16 h state
  __shared__ float accO[16][2];

  const int lane = threadIdx.x & 31;
  const int wave = threadIdx.x >> 5;
  const int b0 = blockIdx.x * 16;
  const int ncol = lane & 15;
  const int mbase = (lane & 16) ? 8 : 0;
  const int hi4 = (lane & 16) ? 4 : 0;

  // h(0) = 0, c(0) = 0, output accumulator = 0.
  for (int i = threadIdx.x; i < 16 * (H_ / 2); i += 256)
    ((unsigned int*)hl)[i] = 0u;
  if (threadIdx.x < 32) ((float*)accO)[threadIdx.x] = 0.0f;
  __syncthreads();

  float cst[8][8];
#pragma unroll
  for (int j = 0; j < 8; ++j)
#pragma unroll
    for (int v = 0; v < 8; ++v) cst[j][v] = 0.0f;

  float bias[8], wo0[8], wo1[8];
#pragma unroll
  for (int j = 0; j < 8; ++j) {
    const int n = wave * 128 + j * 16 + ncol;
    bias[j] = bh[n];
    wo0[j] = Wo[n * 2 + 0];
    wo1[j] = Wo[n * 2 + 1];
  }

  const float bo0 = bo[0], bo1 = bo[1];
  const unsigned int* arow = (const unsigned int*)hl + (lane & 15) * (H_ / 2);

  for (int t = 0; t < L_; ++t) {
    // --- accumulator init: bh + xp[t] (packed C-fragment) -----------------
    const int rt = t * 8 + (b0 >> 4);
    v8f acc[8];
#pragma unroll
    for (int j = 0; j < 8; ++j) {
      const U4 p = xpf[((size_t)rt * 64 + (wave * 8 + j)) * 32 + lane];
      acc[j][0] = bias[j] + bf2f_lo(p.x); acc[j][1] = bias[j] + bf2f_hi(p.x);
      acc[j][2] = bias[j] + bf2f_lo(p.y); acc[j][3] = bias[j] + bf2f_hi(p.y);
      acc[j][4] = bias[j] + bf2f_lo(p.z); acc[j][5] = bias[j] + bf2f_hi(p.z);
      acc[j][6] = bias[j] + bf2f_lo(p.w); acc[j][7] = bias[j] + bf2f_hi(p.w);
    }

    // --- z += h @ Wh : 32 ktiles x 8 ntiles of bf16 WMMA ------------------
#pragma unroll 2
    for (int kt = 0; kt < 32; ++kt) {
      FragU a;
      const int d0 = (kt << 4) + hi4;
      a.q[0] = *(const U4*)(arow + d0);
      a.q[1] = *(const U4*)(arow + d0 + 8);
#pragma unroll
      for (int j = 0; j < 8; ++j) {
        FragU bfr;
        const U4* fb = whb + ((size_t)(kt * 64 + wave * 8 + j)) * 64;
        bfr.q[0] = fb[lane];
        bfr.q[1] = fb[32 + lane];
        acc[j] = __builtin_amdgcn_wmma_f32_16x16x32_bf16(
            false, a.bf, false, bfr.bf, (short)0, acc[j], false, false);
      }
    }
    __syncthreads();  // all waves done reading h(t) from LDS

    // --- elementwise update, h writeback, fused out-proj partials ---------
    float po0[8], po1[8];
#pragma unroll
    for (int v = 0; v < 8; ++v) { po0[v] = 0.0f; po1[v] = 0.0f; }
#pragma unroll
    for (int j = 0; j < 8; ++j) {
      const int n = wave * 128 + j * 16 + ncol;
#pragma unroll
      for (int v = 0; v < 8; ++v) {
        const float z = acc[j][v];
        const float sg = sigmoidf_(z);
        const float g = tanhf(z);
        const float c = sg * (cst[j][v] + g);  // sg*c + sg*g
        cst[j][v] = c;
        const float h = sg * tanhf(c);
        hl[(mbase + v) * H_ + n] = f2bf(h);
        po0[v] += h * wo0[j];
        po1[v] += h * wo1[j];
      }
    }
#pragma unroll
    for (int v = 0; v < 8; ++v) {
      atomicAdd(&accO[mbase + v][0], po0[v]);
      atomicAdd(&accO[mbase + v][1], po1[v]);
    }
    __syncthreads();  // h(t+1) + accO complete

    // --- emit output row (b, t), apply slen mask, reset accumulator -------
    if (threadIdx.x < 16) {
      const int m = threadIdx.x;
      const int b = b0 + m;
      float o0, o1;
      if (t > slen[b]) {
        o0 = 0.0f; o1 = 1.0f;
      } else {
        o0 = sigmoidf_(accO[m][0] + bo0);
        o1 = sigmoidf_(accO[m][1] + bo1);
      }
      const size_t oi = ((size_t)b * L_ + t) * 2;
      out[oi] = o0;
      out[oi + 1] = o1;
      accO[m][0] = 0.0f;
      accO[m][1] = 0.0f;
    }
    __syncthreads();  // accO re-zeroed before next step's adds
  }
}

// ---------------------------------------------------------------------------
extern "C" void kernel_launch(void* const* d_in, const int* in_sizes, int n_in,
                              void* d_out, int out_size, void* d_ws, size_t ws_size,
                              hipStream_t stream) {
  (void)in_sizes; (void)n_in; (void)out_size; (void)ws_size;
  const float* x    = (const float*)d_in[0];
  const int*   slen = (const int*)d_in[1];
  const float* Wh   = (const float*)d_in[2];
  const float* bh   = (const float*)d_in[3];
  const float* Wx   = (const float*)d_in[4];
  const float* bx   = (const float*)d_in[5];
  const float* Wo   = (const float*)d_in[6];
  const float* bo   = (const float*)d_in[7];
  float* out = (float*)d_out;

  char* ws = (char*)d_ws;
  U4* whb = (U4*)(ws);                          // 2 MB: Wh fragments
  U4* wxb = (U4*)(ws + ((size_t)2 << 20));      // 1 MB: Wx fragments
  U4* xpf = (U4*)(ws + ((size_t)3 << 20));      // 64 MB: xp bf16 C-fragments

  rnn_prep_frags<<<dim3(3072), dim3(32), 0, stream>>>(Wh, Wx, whb, wxb);
  rnn_xp_gemm<<<dim3(2048), dim3(256), 0, stream>>>(x, wxb, bx, xpf);
  rnn_scan<<<dim3(8), dim3(256), 0, stream>>>(whb, xpf, bh, Wo, bo, slen, out);
}
